// OnlineCrossAttnQuerySFC2D_85950885528386
// MI455X (gfx1250) — matrix operand order repaired and checked by reading, not compiled, not run on hardware
//
#include <hip/hip_runtime.h>

// ---------------------------------------------------------------------------
// OnlineCrossAttnQuerySFC2D fused kernel for MI455X (gfx1250, wave32, WMMA)
// B=8, C=128, T=128, F=512, K=64.  bf16 WMMA 16x16x32, f32 accumulation.
// One workgroup = (b, t, F-chunk of 256). 256 threads = 8 waves, each wave
// owns two 16-column f-tiles end-to-end. Weights pre-converted to bf16 in
// d_ws, staged per-pass into LDS. Peak LDS = 288 KB (< 320 KB/WGP).
// Round 3: hoist tile-invariant WMMA operands into registers (halves LDS
// fragment traffic); fast v_rcp/v_rsq; vectorized b128 bf16 LDS IO.
// ---------------------------------------------------------------------------

typedef __bf16 bf16_t;
typedef __bf16 v16bf __attribute__((ext_vector_type(16)));
typedef __bf16 v8bf  __attribute__((ext_vector_type(8)));
typedef float  v8f   __attribute__((ext_vector_type(8)));

#define BB 8
#define CC 128
#define TT 128
#define FF 512
#define KK 64
#define FCHUNK 256
#define EPSR 1e-6f

#define DEV static __device__ __forceinline__

DEV float fast_rcp(float x)  { return __builtin_amdgcn_rcpf(x); }   // v_rcp_f32
DEV float fast_rsq(float x)  { return __builtin_amdgcn_rsqf(x); }   // v_rsq_f32
DEV float silu_f(float x)    { return x * fast_rcp(1.f + __expf(-x)); }

// A fragment (16x32 bf16, MxK) from row-major [m][k] storage.
// lane<16: m=lane, k=kb+0..15 ; lane>=16: m=lane-16, k=kb+16..31 (contiguous)
DEV v16bf load_a(const bf16_t* base, int lda, int row0, int kbase, int lane) {
    const int half = lane >> 4;
    const int m    = row0 + (lane & 15);
    const bf16_t* p = base + m * lda + kbase + half * 16;
    v16bf a;
    ((float4*)&a)[0] = *(const float4*)(p);
    ((float4*)&a)[1] = *(const float4*)(p + 8);
    return a;
}

// B fragment (32x16 bf16, KxN) from K-transposed [n][k] storage.
// lane<16: n=lane, k=kb+0..15 ; lane>=16: n=lane-16, k=kb+16..31 (contiguous)
DEV v16bf load_b(const bf16_t* base, int ldk, int col0, int kbase, int lane) {
    const int n  = col0 + (lane & 15);
    const int kb = kbase + (lane >> 4) * 16;
    const bf16_t* p = base + n * ldk + kb;
    v16bf b;
    ((float4*)&b)[0] = *(const float4*)(p);
    ((float4*)&b)[1] = *(const float4*)(p + 8);
    return b;
}

DEV void load_a4(v16bf* dst, const bf16_t* base, int lda, int row0, int lane) {
#pragma unroll
    for (int s = 0; s < 4; ++s) dst[s] = load_a(base, lda, row0, s * 32, lane);
}
DEV void load_b4(v16bf* dst, const bf16_t* base, int ldk, int col0, int lane) {
#pragma unroll
    for (int s = 0; s < 4; ++s) dst[s] = load_b(base, ldk, col0, s * 32, lane);
}

DEV v8f wmma_bf16(v16bf a, v16bf b, v8f c) {
    return __builtin_amdgcn_wmma_f32_16x16x32_bf16(false, a, false, b,
                                                   (short)0, c, false, false);
}

// GEMM with pre-loaded (tile-invariant) B fragments; A streamed from LDS
DEV v8f gemm_pb(const bf16_t* A, int lda, int row0, const v16bf* bfr,
                v8f acc, int lane) {
#pragma unroll
    for (int s = 0; s < 4; ++s)
        acc = wmma_bf16(load_a(A, lda, row0, s * 32, lane), bfr[s], acc);
    return acc;
}
// GEMM with pre-loaded (tile-invariant) A fragments; B streamed from LDS
DEV v8f gemm_pa(const v16bf* afr, const bf16_t* Bm, int ldb, int col0,
                v8f acc, int lane) {
#pragma unroll
    for (int s = 0; s < 4; ++s)
        acc = wmma_bf16(afr[s], load_b(Bm, ldb, col0, s * 32, lane), acc);
    return acc;
}

// acc[r] corresponds to row m = row0 + 8*(lane>=16) + r
DEV v8f bias_acc(const float* __restrict__ bias, int row0, int lane) {
    const int mo = row0 + ((lane >> 4) << 3);
    v8f c;
#pragma unroll
    for (int r = 0; r < 8; ++r) c[r] = bias[mo + r];
    return c;
}

DEV v8bf pack_bf8(v8f acc) {
    v8bf pk;
#pragma unroll
    for (int r = 0; r < 8; ++r) pk[r] = (bf16_t)acc[r];
    return pk;
}

// store D tile transposed into [n][m] storage as one ds_store_b128 per lane
DEV void store_dT(bf16_t* base, int ldm, int row0, int col0, v8f acc, int lane) {
    const int n  = col0 + (lane & 15);
    const int mo = row0 + ((lane >> 4) << 3);
    *(v8bf*)(base + n * ldm + mo) = pack_bf8(acc);
}

// cooperative bf16 weight staging global(L2) -> LDS, 16B per thread per step
DEV void stage_w(const bf16_t* __restrict__ g, bf16_t* l, int n, int tid) {
    for (int i = tid * 8; i < n; i += 256 * 8)
        *(float4*)(l + i) = *(const float4*)(g + i);
}

// ---------------------------------------------------------------------------
__global__ void cvt_bf16_kernel(const float* __restrict__ s,
                                bf16_t* __restrict__ d, int n) {
    int i = blockIdx.x * blockDim.x + threadIdx.x;
    for (; i < n; i += gridDim.x * blockDim.x) d[i] = (bf16_t)s[i];
}

// ---------------------------------------------------------------------------
// LDS map (dynamic shared, 294912 B):
//   [0      .. 98303 ]  wbuf : staged weights for current pass (max 96 KB)
//   [98304  .. 114687]  kT   : k as [k=64][c=128] bf16 (B operand of scores)
//   [114688 .. 131071]  vB   : v as [c=128][k=64] bf16 (A operand of attend)
//   [131072 .. 196607]  qh   : query_h / hidden as [fchunk=256][c=128] bf16
//                              (also latT/lathT scratch during pass 1)
//   [196608 .. 294911]  per-wave scratch: 8 waves x 12 KB (bufA/bufB/bufC)
// ---------------------------------------------------------------------------
__global__ __launch_bounds__(256)
void fused_attn_kernel(
    const float* __restrict__ latent, const float* __restrict__ side,
    const float* __restrict__ basis,
    const float* __restrict__ lp_gamma, const float* __restrict__ lp_b,
    const float* __restrict__ qn_gamma,
    const float* __restrict__ qin_b, const float* __restrict__ qout_b,
    const float* __restrict__ q_b, const float* __restrict__ k_b,
    const float* __restrict__ v_b, const float* __restrict__ o_b,
    const float* __restrict__ ffn_gamma,
    const float* __restrict__ fin_b, const float* __restrict__ fout_b,
    const bf16_t* __restrict__ w_lp, const bf16_t* __restrict__ w_qin,
    const bf16_t* __restrict__ w_qout, const bf16_t* __restrict__ w_q,
    const bf16_t* __restrict__ w_k, const bf16_t* __restrict__ w_v,
    const bf16_t* __restrict__ w_o, const bf16_t* __restrict__ w_fin,
    const bf16_t* __restrict__ w_fout,
    const float* __restrict__ p_sscale, const float* __restrict__ p_pscale,
    const float* __restrict__ p_qss,
    float* __restrict__ out)
{
    extern __shared__ char smem[];
    bf16_t* wbuf = (bf16_t*)(smem);
    bf16_t* kT   = (bf16_t*)(smem + 98304);
    bf16_t* vB   = (bf16_t*)(smem + 114688);
    bf16_t* qh   = (bf16_t*)(smem + 131072);
    char*   scr  = smem + 196608;

    const int tid  = threadIdx.x;
    const int wv   = tid >> 5;
    const int lane = tid & 31;
    const int fblk = blockIdx.x;
    const int t    = blockIdx.y;
    const int b    = blockIdx.z;
    const int f0c  = fblk * FCHUNK;

    bf16_t* bufA = (bf16_t*)(scr + wv * 12288);          // 4 KB
    bf16_t* bufB = (bf16_t*)(scr + wv * 12288 + 4096);   // 4 KB
    bf16_t* bufC = (bf16_t*)(scr + wv * 12288 + 8192);   // 4 KB

    // ================= Pass 1: latent path -> kT, vB =================
    stage_w(w_lp, wbuf,               CC * CC, tid);
    stage_w(w_k,  wbuf + CC * CC,     CC * CC, tid);
    stage_w(w_v,  wbuf + 2 * CC * CC, CC * CC, tid);

    bf16_t* latT  = qh;             // rmsnorm(latent) as [k=64][c=128]
    bf16_t* lathT = qh + 64 * 128;  // latent_h as [k=64][c=128]
    {
        float* red = (float*)scr;   // [256] partials + [64] rstd
        const int kcol = tid & 63;
        const int quad = tid >> 6;
        float lv[32];
        float ss = 0.f;
#pragma unroll
        for (int r = 0; r < 32; ++r) {
            const int c = quad * 32 + r;
            const float x = latent[((b * CC + c) * TT + t) * KK + kcol];
            lv[r] = x; ss += x * x;
        }
        red[tid] = ss;
        __syncthreads();   // also guarantees wbuf staging is complete
        if (tid < 64) {
            const float s = red[tid] + red[tid + 64] + red[tid + 128] + red[tid + 192];
            red[256 + tid] = fast_rsq(s * (1.f / CC) + EPSR);
        }
        __syncthreads();
        const float rs = red[256 + kcol];
#pragma unroll
        for (int rb = 0; rb < 32; rb += 8) {
            v8bf pk;
#pragma unroll
            for (int r2 = 0; r2 < 8; ++r2) {
                const int c = quad * 32 + rb + r2;
                pk[r2] = (bf16_t)(lv[rb + r2] * rs * lp_gamma[c]);
            }
            *(v8bf*)(latT + kcol * CC + quad * 32 + rb) = pk;
        }
    }
    __syncthreads();

    // latent_h = silu(W_lp @ latn + lp_b) ; wave wv owns M-tile wv
    {
        v16bf a4[4];
        load_a4(a4, wbuf, CC, wv * 16, lane);
        for (int nt = 0; nt < 4; ++nt) {
            v8f acc = bias_acc(lp_b, wv * 16, lane);
            acc = gemm_pa(a4, latT, CC, nt * 16, acc, lane);
#pragma unroll
            for (int r = 0; r < 8; ++r) acc[r] = silu_f(acc[r]);
            store_dT(lathT, CC, wv * 16, nt * 16, acc, lane);
        }
    }
    __syncthreads();

    // k = W_k @ latent_h + k_b -> kT[k][c] ; v = W_v @ latent_h + v_b -> vB[c][k]
    {
        v16bf ak[4], av[4];
        load_a4(ak, wbuf + CC * CC,     CC, wv * 16, lane);
        load_a4(av, wbuf + 2 * CC * CC, CC, wv * 16, lane);
        for (int nt = 0; nt < 4; ++nt) {
            v16bf b4[4];
            load_b4(b4, lathT, CC, nt * 16, lane);
            v8f acck = bias_acc(k_b, wv * 16, lane);
            v8f accv = bias_acc(v_b, wv * 16, lane);
#pragma unroll
            for (int s = 0; s < 4; ++s) {
                acck = wmma_bf16(ak[s], b4[s], acck);
                accv = wmma_bf16(av[s], b4[s], accv);
            }
            store_dT(kT, CC, wv * 16, nt * 16, acck, lane);
            const int n  = nt * 16 + (lane & 15);
            const int mo = wv * 16 + ((lane >> 4) << 3);
#pragma unroll
            for (int r = 0; r < 8; ++r) vB[(mo + r) * KK + n] = (bf16_t)accv[r];
        }
    }
    __syncthreads();

    // ================= Pass 2: query_h (SwiGLU MLP) =================
    stage_w(w_qin,  wbuf,              2 * CC * CC, tid);
    stage_w(w_qout, wbuf + 2 * CC * CC,    CC * CC, tid);
    __syncthreads();

    for (int ft = 2 * wv; ft < 2 * wv + 2; ++ft) {
        const int f0 = f0c + ft * 16;
        const int j  = lane & 15, hh = lane >> 4;

        // per-column RMSNorm of side (2 lanes/column, shuffle-combined)
        float sv[64]; float ss = 0.f;
#pragma unroll
        for (int r = 0; r < 64; ++r) {
            const int c = hh * 64 + r;
            const float x = side[((b * CC + c) * TT + t) * FF + f0 + j];
            sv[r] = x; ss += x * x;
        }
        ss += __shfl_xor(ss, 16, 32);
        const float rs = fast_rsq(ss * (1.f / CC) + EPSR);
#pragma unroll
        for (int rb = 0; rb < 64; rb += 8) {
            v8bf pk;
#pragma unroll
            for (int r2 = 0; r2 < 8; ++r2) {
                const int c = hh * 64 + rb + r2;
                pk[r2] = (bf16_t)(sv[rb + r2] * rs * qn_gamma[c]);
            }
            *(v8bf*)(bufA + j * CC + hh * 64 + rb) = pk;
        }

        // h = W_qin @ xn ; u = a * silu(g)  (a/g M-tile pairs, B hoisted)
        {
            v16bf b4[4];
            load_b4(b4, bufA, CC, 0, lane);
            for (int mt = 0; mt < 8; ++mt) {
                v8f acc_a = bias_acc(qin_b, mt * 16, lane);
                v8f acc_g = bias_acc(qin_b, CC + mt * 16, lane);
#pragma unroll
                for (int s = 0; s < 4; ++s) {
                    acc_a = wmma_bf16(load_a(wbuf, CC, mt * 16, s * 32, lane), b4[s], acc_a);
                    acc_g = wmma_bf16(load_a(wbuf, CC, CC + mt * 16, s * 32, lane), b4[s], acc_g);
                }
                const int mo = mt * 16 + (hh << 3);
                v8bf pk;
#pragma unroll
                for (int r = 0; r < 8; ++r)
                    pk[r] = (bf16_t)(acc_a[r] * silu_f(acc_g[r]));
                *(v8bf*)(bufB + j * CC + mo) = pk;
            }
        }
        // query_h = W_qout @ u + b -> qh[f][c]  (B hoisted)
        {
            v16bf u4[4];
            load_b4(u4, bufB, CC, 0, lane);
            for (int mt = 0; mt < 8; ++mt) {
                v8f acc = bias_acc(qout_b, mt * 16, lane);
                acc = gemm_pb(wbuf + 2 * CC * CC, CC, mt * 16, u4, acc, lane);
                store_dT(qh, CC, mt * 16, ft * 16, acc, lane);
            }
        }
    }
    __syncthreads();

    // ================= Pass 3: q, scores, softmax, attend, o-proj ===
    stage_w(w_q, wbuf,           CC * CC, tid);
    stage_w(w_o, wbuf + CC * CC, CC * CC, tid);
    __syncthreads();
    const float sscale = p_sscale[0], pscale = p_pscale[0], qss = p_qss[0];

    for (int ft = 2 * wv; ft < 2 * wv + 2; ++ft) {
        const int f0 = f0c + ft * 16;
        const int ln = lane & 15, hh = lane >> 4;

        // q = W_q @ query_h + q_b -> bufA[f][c]  (B = query_h cols, hoisted)
        {
            v16bf qh4[4];
            load_b4(qh4, qh, CC, ft * 16, lane);
            for (int mt = 0; mt < 8; ++mt) {
                v8f acc = bias_acc(q_b, mt * 16, lane);
                acc = gemm_pb(wbuf, CC, mt * 16, qh4, acc, lane);
                store_dT(bufA, CC, mt * 16, 0, acc, lane);
            }
        }
        // scores[f][k] = (q^T k)*scale + basis[k][f]*prior  (A = q^T, hoisted)
        v8f sc[4];
        {
            v16bf qa4[4];
            load_a4(qa4, bufA, CC, 0, lane);
            for (int nt = 0; nt < 4; ++nt) {
                v8f z = {0.f, 0.f, 0.f, 0.f, 0.f, 0.f, 0.f, 0.f};
                sc[nt] = gemm_pa(qa4, kT, CC, nt * 16, z, lane);
            }
        }
#pragma unroll
        for (int nt = 0; nt < 4; ++nt)
#pragma unroll
            for (int r = 0; r < 8; ++r) {
                const int kcol = nt * 16 + ln;
                const int fg   = f0 + r + (hh << 3);
                sc[nt][r] = sc[nt][r] * sscale + pscale * basis[kcol * FF + fg];
            }
        // softmax over k: reduce over 4 N-tiles and across 16-lane half
#pragma unroll
        for (int r = 0; r < 8; ++r) {
            float mx = fmaxf(fmaxf(sc[0][r], sc[1][r]), fmaxf(sc[2][r], sc[3][r]));
#pragma unroll
            for (int m = 1; m <= 8; m <<= 1) mx = fmaxf(mx, __shfl_xor(mx, m, 32));
            float sum = 0.f;
#pragma unroll
            for (int nt = 0; nt < 4; ++nt) {
                const float e = __expf(sc[nt][r] - mx);
                sc[nt][r] = e; sum += e;
            }
#pragma unroll
            for (int m = 1; m <= 8; m <<= 1) sum += __shfl_xor(sum, m, 32);
            const float inv = fast_rcp(sum);
            const int fl = r + (hh << 3);
#pragma unroll
            for (int nt = 0; nt < 4; ++nt)
                bufC[fl * KK + nt * 16 + ln] = (bf16_t)(sc[nt][r] * inv);
        }
        // attended = v @ w^T -> bufA[f][c]  (B = softmax weights, hoisted)
        {
            v16bf w2[2];
#pragma unroll
            for (int s = 0; s < 2; ++s) w2[s] = load_b(bufC, KK, 0, s * 32, lane);
            for (int mt = 0; mt < 8; ++mt) {
                v8f acc = {0.f, 0.f, 0.f, 0.f, 0.f, 0.f, 0.f, 0.f};
#pragma unroll
                for (int s = 0; s < 2; ++s)
                    acc = wmma_bf16(load_a(vB, KK, mt * 16, s * 32, lane), w2[s], acc);
                store_dT(bufA, CC, mt * 16, 0, acc, lane);
            }
        }
        // hidden = W_o @ attended + o_b + qss*query_h -> qh (aliased, own cols)
        {
            v16bf at4[4];
            load_b4(at4, bufA, CC, 0, lane);
            for (int mt = 0; mt < 8; ++mt) {
                v8f acc = bias_acc(o_b, mt * 16, lane);
                acc = gemm_pb(wbuf + CC * CC, CC, mt * 16, at4, acc, lane);
                const int n  = ft * 16 + ln;
                const int mo = mt * 16 + (hh << 3);
                bf16_t* hp = qh + n * CC + mo;
                const v8bf hold = *(const v8bf*)hp;
                v8bf pk;
#pragma unroll
                for (int r = 0; r < 8; ++r)
                    pk[r] = (bf16_t)(acc[r] + qss * (float)hold[r]);
                *(v8bf*)hp = pk;
            }
        }
    }
    __syncthreads();

    // ================= Pass 4: FFN (SwiGLU) + residual -> out =======
    stage_w(w_fin,  wbuf,              2 * CC * CC, tid);
    stage_w(w_fout, wbuf + 2 * CC * CC,    CC * CC, tid);
    __syncthreads();

    for (int ft = 2 * wv; ft < 2 * wv + 2; ++ft) {
        const int f0 = f0c + ft * 16;
        const int j  = lane & 15, hh = lane >> 4;

        // RMSNorm of hidden column (vector bf16 reads from qh[f][c])
        float hv[64]; float ss = 0.f;
#pragma unroll
        for (int rb = 0; rb < 64; rb += 8) {
            const v8bf hb = *(const v8bf*)(qh + (ft * 16 + j) * CC + hh * 64 + rb);
#pragma unroll
            for (int r2 = 0; r2 < 8; ++r2) {
                const float x = (float)hb[r2];
                hv[rb + r2] = x; ss += x * x;
            }
        }
        ss += __shfl_xor(ss, 16, 32);
        const float rs = fast_rsq(ss * (1.f / CC) + EPSR);
#pragma unroll
        for (int rb = 0; rb < 64; rb += 8) {
            v8bf pk;
#pragma unroll
            for (int r2 = 0; r2 < 8; ++r2) {
                const int c = hh * 64 + rb + r2;
                pk[r2] = (bf16_t)(hv[rb + r2] * rs * ffn_gamma[c]);
            }
            *(v8bf*)(bufA + j * CC + hh * 64 + rb) = pk;
        }
        // ffn_in + swiglu -> bufB  (B hoisted)
        {
            v16bf b4[4];
            load_b4(b4, bufA, CC, 0, lane);
            for (int mt = 0; mt < 8; ++mt) {
                v8f acc_a = bias_acc(fin_b, mt * 16, lane);
                v8f acc_g = bias_acc(fin_b, CC + mt * 16, lane);
#pragma unroll
                for (int s = 0; s < 4; ++s) {
                    acc_a = wmma_bf16(load_a(wbuf, CC, mt * 16, s * 32, lane), b4[s], acc_a);
                    acc_g = wmma_bf16(load_a(wbuf, CC, CC + mt * 16, s * 32, lane), b4[s], acc_g);
                }
                const int mo = mt * 16 + (hh << 3);
                v8bf pk;
#pragma unroll
                for (int r = 0; r < 8; ++r)
                    pk[r] = (bf16_t)(acc_a[r] * silu_f(acc_g[r]));
                *(v8bf*)(bufB + j * CC + mo) = pk;
            }
        }
        // out = hidden + W_fout @ u + fout_b  (B hoisted)
        {
            v16bf u4[4];
            load_b4(u4, bufB, CC, 0, lane);
            for (int mt = 0; mt < 8; ++mt) {
                v8f acc = bias_acc(fout_b, mt * 16, lane);
                acc = gemm_pb(wbuf + 2 * CC * CC, CC, mt * 16, u4, acc, lane);
                const int mo = mt * 16 + (hh << 3);
                const int n  = ft * 16 + j;
                const v8bf hb = *(const v8bf*)(qh + n * CC + mo);
#pragma unroll
                for (int r = 0; r < 8; ++r)
                    out[((b * CC + mo + r) * TT + t) * FF + f0 + j] = acc[r] + (float)hb[r];
            }
        }
    }
}

// ---------------------------------------------------------------------------
extern "C" void kernel_launch(void* const* d_in, const int* in_sizes, int n_in,
                              void* d_out, int out_size, void* d_ws, size_t ws_size,
                              hipStream_t stream) {
    const float* latent    = (const float*)d_in[0];
    const float* side      = (const float*)d_in[1];
    const float* basis     = (const float*)d_in[2];
    const float* lp_gamma  = (const float*)d_in[3];
    const float* lp_w      = (const float*)d_in[4];
    const float* lp_b      = (const float*)d_in[5];
    const float* qn_gamma  = (const float*)d_in[6];
    const float* qin_w     = (const float*)d_in[7];
    const float* qin_b     = (const float*)d_in[8];
    const float* qout_w    = (const float*)d_in[9];
    const float* qout_b    = (const float*)d_in[10];
    const float* q_w       = (const float*)d_in[11];
    const float* q_b       = (const float*)d_in[12];
    const float* k_w       = (const float*)d_in[13];
    const float* k_b       = (const float*)d_in[14];
    const float* v_w       = (const float*)d_in[15];
    const float* v_b       = (const float*)d_in[16];
    const float* o_w       = (const float*)d_in[17];
    const float* o_b       = (const float*)d_in[18];
    const float* ffn_gamma = (const float*)d_in[19];
    const float* fin_w     = (const float*)d_in[20];
    const float* fin_b     = (const float*)d_in[21];
    const float* fout_w    = (const float*)d_in[22];
    const float* fout_b    = (const float*)d_in[23];
    const float* sscale    = (const float*)d_in[24];
    const float* pscale    = (const float*)d_in[25];
    const float* qss       = (const float*)d_in[26];
    float* out = (float*)d_out;

    // bf16 weight cache in workspace (element offsets)
    bf16_t* ws     = (bf16_t*)d_ws;
    bf16_t* b_lp   = ws;
    bf16_t* b_qin  = ws + 16384;
    bf16_t* b_qout = ws + 49152;
    bf16_t* b_q    = ws + 65536;
    bf16_t* b_k    = ws + 81920;
    bf16_t* b_v    = ws + 98304;
    bf16_t* b_o    = ws + 114688;
    bf16_t* b_fin  = ws + 131072;
    bf16_t* b_fout = ws + 163840;

    const int cc = CC * CC, cc2 = 2 * CC * CC;
    cvt_bf16_kernel<<<dim3(64), dim3(256), 0, stream>>>(lp_w,   b_lp,   cc);
    cvt_bf16_kernel<<<dim3(128), dim3(256), 0, stream>>>(qin_w, b_qin,  cc2);
    cvt_bf16_kernel<<<dim3(64), dim3(256), 0, stream>>>(qout_w, b_qout, cc);
    cvt_bf16_kernel<<<dim3(64), dim3(256), 0, stream>>>(q_w,    b_q,    cc);
    cvt_bf16_kernel<<<dim3(64), dim3(256), 0, stream>>>(k_w,    b_k,    cc);
    cvt_bf16_kernel<<<dim3(64), dim3(256), 0, stream>>>(v_w,    b_v,    cc);
    cvt_bf16_kernel<<<dim3(64), dim3(256), 0, stream>>>(o_w,    b_o,    cc);
    cvt_bf16_kernel<<<dim3(128), dim3(256), 0, stream>>>(fin_w, b_fin,  cc2);
    cvt_bf16_kernel<<<dim3(64), dim3(256), 0, stream>>>(fout_w, b_fout, cc);

    const int lds_bytes = 294912;  // 288 KB dynamic LDS
    (void)hipFuncSetAttribute((const void*)fused_attn_kernel,
                              hipFuncAttributeMaxDynamicSharedMemorySize, lds_bytes);

    fused_attn_kernel<<<dim3(FF / FCHUNK, TT, BB), dim3(256), lds_bytes, stream>>>(
        latent, side, basis,
        lp_gamma, lp_b, qn_gamma, qin_b, qout_b, q_b, k_b, v_b, o_b,
        ffn_gamma, fin_b, fout_b,
        b_lp, b_qin, b_qout, b_q, b_k, b_v, b_o, b_fin, b_fout,
        sscale, pscale, qss, out);
}
